// Attention_70557722739202
// MI455X (gfx1250) — matrix operand order repaired and checked by reading, not compiled, not run on hardware
//
#include <hip/hip_runtime.h>
#include <stdint.h>

// ---------------------------------------------------------------------------
// Types for CDNA5 WMMA (wave32, 16x16x32 bf16 -> f32)
// ---------------------------------------------------------------------------
typedef __bf16 bf16;
typedef __attribute__((ext_vector_type(16))) __bf16 v16bf;
typedef __attribute__((ext_vector_type(8)))  __bf16 v8bf;
typedef __attribute__((ext_vector_type(4)))  __bf16 v4bf;
typedef __attribute__((ext_vector_type(8)))  float  v8f;
typedef __attribute__((ext_vector_type(4)))  float  v4f;
typedef __attribute__((ext_vector_type(4)))  unsigned int v4u;
typedef __attribute__((ext_vector_type(8)))  int    v8i;
typedef __attribute__((ext_vector_type(4)))  int    v4i;

static constexpr int DIM  = 768;
static constexpr int NH   = 12;
static constexpr int HD   = 64;
static constexpr int SEQ  = 2048;
static constexpr int BATCH= 4;
static constexpr int MTOT = BATCH * SEQ;        // 8192 rows
static constexpr int QKVN = 3 * DIM;            // 2304

__device__ __forceinline__ v8f wmma_bf16(v16bf a, v16bf b, v8f c) {
    return __builtin_amdgcn_wmma_f32_16x16x32_bf16(false, a, false, b,
                                                   (short)0, c, false, false);
}

// CDNA5 async copy: global -> LDS without VGPR round-trip (ASYNCcnt tracked).
__device__ __forceinline__ void async_b128(void* lds_ptr, const void* gptr) {
    uint32_t lo = (uint32_t)(uintptr_t)lds_ptr;
    asm volatile("global_load_async_to_lds_b128 %0, %1, off"
                 :: "v"(lo), "v"((uint64_t)(uintptr_t)gptr) : "memory");
}
__device__ __forceinline__ void wait_async0() {
    asm volatile("s_wait_asynccnt 0x0" ::: "memory");
}

// ---------------------------------------------------------------------------
// CDNA5 Tensor Data Mover: 2D tile (tile_d1 rows x tile_d0 elems, bf16) from
// global (row stride stride_d0 elems) into LDS, packed row-major.
// D# layout per ISA 8.3/8.4: group0 = {count, lds_addr, global_addr, type=2},
// group1 = {data_size=2B, tensor dims (large, no OOB), tile dims, stride}.
// This toolchain exposes the 6-arg builtin:
//   (uint32x4 g0, int32x8 g1, int32x4, int32x4, int32x8, i32 cpol)
// ---------------------------------------------------------------------------
__device__ __forceinline__ void tdm_load_2d(void* lds_ptr, const void* gptr,
                                            uint32_t tile_d0, uint32_t tile_d1,
                                            uint32_t stride_d0) {
    uint64_t ga = (uint64_t)(uintptr_t)gptr;
    v4u g0;
    g0[0] = 1u;                                        // count=1 (user descriptor)
    g0[1] = (uint32_t)(uintptr_t)lds_ptr;              // lds_addr (bytes)
    g0[2] = (uint32_t)(ga & 0xffffffffu);              // global_addr[31:0]
    g0[3] = (uint32_t)((ga >> 32) & 0x01ffffffu)       // global_addr[56:32]
          | 0x80000000u;                               // type=2 ("image")
    const uint32_t td0 = 0x100000u, td1 = 0x100000u;   // tensor dims: large
    v8i g1;
    g1[0] = (int)(1u << 16);                                           // data_size=2B
    g1[1] = (int)((td0 & 0xffffu) << 16);                              // dim0[15:0]
    g1[2] = (int)(((td0 >> 16) & 0xffffu) | ((td1 & 0xffffu) << 16));  // dim0[31:16],dim1[15:0]
    g1[3] = (int)(((td1 >> 16) & 0xffffu) | ((tile_d0 & 0xffffu) << 16)); // tile_dim0
    g1[4] = (int)(tile_d1 & 0xffffu);                                  // tile_dim1, tile_dim2=0
    g1[5] = (int)stride_d0;                                            // dim0_stride[31:0]
    g1[6] = 0;                                                         // stride0[47:32], stride1[15:0]
    g1[7] = 0;
    v4i z4 = (v4i){0, 0, 0, 0};                        // groups 2/3 unused (2D)
    v8i z8 = (v8i){0, 0, 0, 0, 0, 0, 0, 0};
    __builtin_amdgcn_tensor_load_to_lds(g0, g1, z4, z4, z8, 0);
}

// A-fragment (16x32, bf16, row-major source, leading dim ld):
// lane r = lane&15 holds row r; kb = 8*(lane>=16); elems 0..7 -> K kb..kb+7,
// elems 8..15 -> K kb+16..kb+23  (ISA 7.12.2 16-bit A layout)
__device__ __forceinline__ v16bf load_a_frag(const bf16* base, int ld) {
    const int lane = threadIdx.x & 31;
    const int r    = lane & 15;
    const int kb   = (lane >> 4) << 3;
    const bf16* p  = base + r * ld + kb;
    v8bf lo = *(const v8bf*)(p);
    v8bf hi = *(const v8bf*)(p + 16);
    return __builtin_shufflevector(lo, hi, 0,1,2,3,4,5,6,7,8,9,10,11,12,13,14,15);
}

// B-fragment (32x16): B[k][n] = src[n*ld + k]. lane col n = lane&15,
// kb = 8*(lane>=16); elems 0..7 -> K kb..kb+7, elems 8..15 -> K kb+16..kb+23.
__device__ __forceinline__ v16bf load_b_frag(const bf16* base, int ld) {
    const int lane = threadIdx.x & 31;
    const int n    = lane & 15;
    const int kb   = (lane >> 4) << 3;
    const bf16* p  = base + n * ld + kb;
    v8bf lo = *(const v8bf*)(p);
    v8bf hi = *(const v8bf*)(p + 16);
    return __builtin_shufflevector(lo, hi, 0,1,2,3,4,5,6,7,8,9,10,11,12,13,14,15);
}

// ---------------------------------------------------------------------------
// Kernel 0: fp32 -> bf16 conversion
// ---------------------------------------------------------------------------
__global__ void cvt_f32_bf16(const float* __restrict__ src, bf16* __restrict__ dst, int n) {
    int idx = (blockIdx.x * blockDim.x + threadIdx.x) * 4;
    if (idx + 3 < n) {
        v4f f = *(const v4f*)(src + idx);
        v4bf o;
        o[0] = (bf16)f[0]; o[1] = (bf16)f[1]; o[2] = (bf16)f[2]; o[3] = (bf16)f[3];
        *(v4bf*)(dst + idx) = o;
    } else {
        for (int i = idx; i < n; ++i) dst[i] = (bf16)src[i];
    }
}

// ---------------------------------------------------------------------------
// Kernel 1: QKV projection  qkv[m][c] = x[m][:] . Wqkv[c][:] + bqkv[c]
// 256 threads (8 waves), 128x128 tile, K-step 32, double-buffered async LDS.
// ---------------------------------------------------------------------------
__global__ __launch_bounds__(256) void qkv_gemm(
        const bf16* __restrict__ X, const bf16* __restrict__ W,
        const float* __restrict__ bias,
        bf16* __restrict__ q, bf16* __restrict__ k, bf16* __restrict__ vt)
{
    __shared__ bf16 As[2][128][32];   // 16 KB
    __shared__ bf16 Bs[2][128][32];   // 16 KB

    const int tid  = threadIdx.x;
    const int wave = tid >> 5;
    const int lane = tid & 31;
    const int m0   = blockIdx.x * 128;
    const int n0   = blockIdx.y * 128;
    const int wm   = wave & 3;
    const int wn   = wave >> 2;

    v8f acc[2][4];
    for (int a = 0; a < 2; ++a)
        for (int t = 0; t < 4; ++t) acc[a][t] = (v8f){0,0,0,0,0,0,0,0};

    const int ldr = tid >> 1;             // 0..127
    const int ldc = (tid & 1) << 4;       // 0 or 16

    auto stage = [&](int buf, int k0) {
        const bf16* sa = X + (size_t)(m0 + ldr) * DIM + k0 + ldc;
        const bf16* sb = W + (size_t)(n0 + ldr) * DIM + k0 + ldc;
        async_b128(&As[buf][ldr][ldc],     sa);
        async_b128(&As[buf][ldr][ldc + 8], sa + 8);
        async_b128(&Bs[buf][ldr][ldc],     sb);
        async_b128(&Bs[buf][ldr][ldc + 8], sb + 8);
    };

    stage(0, 0);
    int cur = 0;
    for (int k0 = 0; k0 < DIM; k0 += 32, cur ^= 1) {
        wait_async0();
        __syncthreads();
        if (k0 + 32 < DIM) stage(cur ^ 1, k0 + 32);  // overlap DMA with WMMA

        v16bf af0 = load_a_frag(&As[cur][wm * 32][0], 32);
        v16bf af1 = load_a_frag(&As[cur][wm * 32 + 16][0], 32);
        v16bf bfr[4];
        for (int t = 0; t < 4; ++t)                  // pre-load -> staggered dscnt
            bfr[t] = load_b_frag(&Bs[cur][wn * 64 + t * 16][0], 32);
        for (int t = 0; t < 4; ++t) {
            acc[0][t] = wmma_bf16(af0, bfr[t], acc[0][t]);
            acc[1][t] = wmma_bf16(af1, bfr[t], acc[1][t]);
        }
    }

    // epilogue: wave's 64 cols = one (which, head) chunk since 64 | col base
    const int cw    = n0 + wn * 64;           // multiple of 64
    const int which = cw / DIM;               // 0=q 1=k 2=v
    const int h     = (cw % DIM) / HD;
    const int b     = m0 / SEQ;               // 128 | 2048 -> constant per block
    const int nbase = m0 % SEQ;
    const int cl    = lane & 15;
    const int mb    = (lane >> 4) << 3;

    for (int mt = 0; mt < 2; ++mt) {
        const int row0 = nbase + wm * 32 + mt * 16 + mb;  // seq position
        for (int t = 0; t < 4; ++t) {
            const int   d   = t * 16 + cl;
            const float bia = bias[cw + t * 16 + cl];
            v8f c = acc[mt][t];
            if (which == 0) {           // Q: fold softmax scale 1/sqrt(64)
                bf16* dst = q + (((size_t)b * NH + h) * SEQ) * HD;
                for (int i = 0; i < 8; ++i)
                    dst[(size_t)(row0 + i) * HD + d] = (bf16)((c[i] + bia) * 0.125f);
            } else if (which == 1) {    // K row-major (n, d)
                bf16* dst = k + (((size_t)b * NH + h) * SEQ) * HD;
                for (int i = 0; i < 8; ++i)
                    dst[(size_t)(row0 + i) * HD + d] = (bf16)(c[i] + bia);
            } else {                    // V transposed (d, n): contiguous pack
                bf16* dst = vt + (((size_t)b * NH + h) * HD + d) * SEQ;
                v8bf pk;
                for (int i = 0; i < 8; ++i) pk[i] = (bf16)(c[i] + bia);
                *(v8bf*)&dst[row0] = pk;
            }
        }
    }
}

// ---------------------------------------------------------------------------
// Kernel 2: flash attention per (b,h). Block = 128 queries (8 waves x 16).
// 32-key tiles of K (row-major) and V^T (d-major) staged by the Tensor Data
// Mover into double-buffered LDS; wave 0 drives TDM + TENSORcnt.
// ---------------------------------------------------------------------------
__global__ __launch_bounds__(256) void flash_attn(
        const bf16* __restrict__ q, const bf16* __restrict__ kk,
        const bf16* __restrict__ vt, bf16* __restrict__ ob)
{
    __shared__ bf16 Kt[2][32][64];     // 8 KB : keys j..j+31, d 0..63
    __shared__ bf16 Vt[2][64][32];     // 8 KB : d 0..63, keys j..j+31
    __shared__ bf16 Ps[8][16][32];     // 8 KB : per-wave P staging (C -> A)

    const int tid  = threadIdx.x;
    const int wave = tid >> 5;
    const int lane = tid & 31;
    const int bh   = blockIdx.y;                   // 0..47
    const int m0   = blockIdx.x * 128 + wave * 16; // query tile start

    const bf16* qp = q  + (size_t)bh * SEQ * HD;
    const bf16* kp = kk + (size_t)bh * SEQ * HD;
    const bf16* vp = vt + (size_t)bh * HD * SEQ;

    // Q fragments (held for the whole sweep)
    v16bf qa0 = load_a_frag(qp + (size_t)m0 * HD, HD);       // d 0..31
    v16bf qa1 = load_a_frag(qp + (size_t)m0 * HD + 32, HD);  // d 32..63

    v16bf ones;
    for (int i = 0; i < 16; ++i) ones[i] = (bf16)1.0f;

    v8f o[4];
    for (int t = 0; t < 4; ++t) o[t] = (v8f){0,0,0,0,0,0,0,0};
    float mrow[8], lrow[8];
    for (int i = 0; i < 8; ++i) { mrow[i] = -3.0e38f; lrow[i] = 0.0f; }

    const int mb = (lane >> 4) << 3;
    const int cl = lane & 15;

    // TDM staging: K tile = 32 rows x 64 elems (row stride 64),
    //              V tile = 64 rows x 32 elems (row stride SEQ)
    auto stage = [&](int buf, int j) {
        if (wave == 0) {
            tdm_load_2d(&Kt[buf][0][0], kp + (size_t)j * HD, HD, 32, HD);
            tdm_load_2d(&Vt[buf][0][0], vp + j,              32, HD, SEQ);
        }
    };

    stage(0, 0);
    int cur = 0;
    for (int j = 0; j < SEQ; j += 32, cur ^= 1) {
        if (wave == 0) __builtin_amdgcn_s_wait_tensorcnt(0);
        __syncthreads();
        if (j + 32 < SEQ) stage(cur ^ 1, j + 32);  // overlap TDM with compute

        // S = Q K^T (scale pre-folded into Q); pre-load K frags
        v16bf kb00 = load_b_frag(&Kt[cur][0][0],   64);
        v16bf kb01 = load_b_frag(&Kt[cur][0][32],  64);
        v16bf kb10 = load_b_frag(&Kt[cur][16][0],  64);
        v16bf kb11 = load_b_frag(&Kt[cur][16][32], 64);
        v8f s0 = (v8f){0,0,0,0,0,0,0,0}, s1 = s0;
        s0 = wmma_bf16(qa0, kb00, s0);
        s0 = wmma_bf16(qa1, kb01, s0);
        s1 = wmma_bf16(qa0, kb10, s1);
        s1 = wmma_bf16(qa1, kb11, s1);

        // row-max over the 32 keys (rows live inside 16-lane halves)
        float tm[8];
        for (int i = 0; i < 8; ++i) tm[i] = fmaxf(s0[i], s1[i]);
        for (int st = 1; st < 16; st <<= 1)
            for (int i = 0; i < 8; ++i)
                tm[i] = fmaxf(tm[i], __shfl_xor(tm[i], st, 32));

        float alpha[8];
        for (int i = 0; i < 8; ++i) {
            float mn = fmaxf(mrow[i], tm[i]);
            alpha[i] = __expf(mrow[i] - mn);
            mrow[i]  = mn;
            s0[i]    = __expf(s0[i] - mn);
            s1[i]    = __expf(s1[i] - mn);
        }

        // stage P (C layout) into LDS as a 16x32 row-major tile
        for (int i = 0; i < 8; ++i) {
            Ps[wave][mb + i][cl]      = (bf16)s0[i];
            Ps[wave][mb + i][cl + 16] = (bf16)s1[i];
        }
        asm volatile("" ::: "memory");  // keep DS store->load order (in-order DS)
        v16bf pa = load_a_frag(&Ps[wave][0][0], 32);

        // row sums via WMMA against all-ones B (every lane gets its 8 rows)
        v8f ls = (v8f){0,0,0,0,0,0,0,0};
        ls = wmma_bf16(pa, ones, ls);
        for (int i = 0; i < 8; ++i) lrow[i] = lrow[i] * alpha[i] + ls[i];

        // rescale O, then O += P V (pre-load V frags)
        for (int t = 0; t < 4; ++t)
            for (int i = 0; i < 8; ++i) o[t][i] *= alpha[i];
        v16bf vb[4];
        for (int t = 0; t < 4; ++t) vb[t] = load_b_frag(&Vt[cur][t * 16][0], 32);
        for (int t = 0; t < 4; ++t) o[t] = wmma_bf16(pa, vb[t], o[t]);
    }

    // normalize and store (b, n, h*64+d) row-major [8192][768]
    const int b = bh / NH, h = bh % NH;
    const size_t rowbase = (size_t)b * SEQ + (m0 % SEQ) + mb;
    float inv[8];
    for (int i = 0; i < 8; ++i) inv[i] = 1.0f / lrow[i];
    for (int t = 0; t < 4; ++t)
        for (int i = 0; i < 8; ++i)
            ob[(rowbase + i) * DIM + h * HD + t * 16 + cl] = (bf16)(o[t][i] * inv[i]);
}

// ---------------------------------------------------------------------------
// Kernel 3: output projection  out[m][n] = ob[m][:] . Wproj[n][:] + bproj[n]
// ---------------------------------------------------------------------------
__global__ __launch_bounds__(256) void proj_gemm(
        const bf16* __restrict__ X, const bf16* __restrict__ W,
        const float* __restrict__ bias, float* __restrict__ out)
{
    __shared__ bf16 As[2][128][32];
    __shared__ bf16 Bs[2][128][32];

    const int tid  = threadIdx.x;
    const int wave = tid >> 5;
    const int lane = tid & 31;
    const int m0   = blockIdx.x * 128;
    const int n0   = blockIdx.y * 128;
    const int wm   = wave & 3;
    const int wn   = wave >> 2;

    v8f acc[2][4];
    for (int a = 0; a < 2; ++a)
        for (int t = 0; t < 4; ++t) acc[a][t] = (v8f){0,0,0,0,0,0,0,0};

    const int ldr = tid >> 1;
    const int ldc = (tid & 1) << 4;

    auto stage = [&](int buf, int k0) {
        const bf16* sa = X + (size_t)(m0 + ldr) * DIM + k0 + ldc;
        const bf16* sb = W + (size_t)(n0 + ldr) * DIM + k0 + ldc;
        async_b128(&As[buf][ldr][ldc],     sa);
        async_b128(&As[buf][ldr][ldc + 8], sa + 8);
        async_b128(&Bs[buf][ldr][ldc],     sb);
        async_b128(&Bs[buf][ldr][ldc + 8], sb + 8);
    };

    stage(0, 0);
    int cur = 0;
    for (int k0 = 0; k0 < DIM; k0 += 32, cur ^= 1) {
        wait_async0();
        __syncthreads();
        if (k0 + 32 < DIM) stage(cur ^ 1, k0 + 32);

        v16bf af0 = load_a_frag(&As[cur][wm * 32][0], 32);
        v16bf af1 = load_a_frag(&As[cur][wm * 32 + 16][0], 32);
        v16bf bfr[4];
        for (int t = 0; t < 4; ++t)
            bfr[t] = load_b_frag(&Bs[cur][wn * 64 + t * 16][0], 32);
        for (int t = 0; t < 4; ++t) {
            acc[0][t] = wmma_bf16(af0, bfr[t], acc[0][t]);
            acc[1][t] = wmma_bf16(af1, bfr[t], acc[1][t]);
        }
    }

    const int cl = lane & 15;
    const int mb = (lane >> 4) << 3;
    for (int mt = 0; mt < 2; ++mt) {
        const int row0 = m0 + wm * 32 + mt * 16 + mb;
        for (int t = 0; t < 4; ++t) {
            const int   n   = n0 + wn * 64 + t * 16 + cl;
            const float bia = bias[n];
            v8f c = acc[mt][t];
            for (int i = 0; i < 8; ++i)
                out[(size_t)(row0 + i) * DIM + n] = c[i] + bia;
        }
    }
}

// ---------------------------------------------------------------------------
// Host launcher
// ---------------------------------------------------------------------------
extern "C" void kernel_launch(void* const* d_in, const int* in_sizes, int n_in,
                              void* d_out, int out_size, void* d_ws, size_t ws_size,
                              hipStream_t stream) {
    const float* x     = (const float*)d_in[0];
    const float* Wqkv  = (const float*)d_in[1];
    const float* bqkv  = (const float*)d_in[2];
    const float* Wproj = (const float*)d_in[3];
    const float* bproj = (const float*)d_in[4];
    float*       out   = (float*)d_out;

    char*  ws  = (char*)d_ws;
    size_t off = 0;
    auto alloc = [&](size_t elems) {
        void* p = ws + off;
        off += (elems * sizeof(bf16) + 255) & ~(size_t)255;
        return (bf16*)p;
    };
    bf16* xb  = alloc((size_t)MTOT * DIM);     // 8192 x 768
    bf16* wqb = alloc((size_t)QKVN * DIM);     // 2304 x 768
    bf16* wpb = alloc((size_t)DIM * DIM);      // 768 x 768
    bf16* q   = alloc((size_t)BATCH * NH * SEQ * HD);
    bf16* k   = alloc((size_t)BATCH * NH * SEQ * HD);
    bf16* vt  = alloc((size_t)BATCH * NH * HD * SEQ);
    bf16* ob  = alloc((size_t)MTOT * DIM);     // attention output, row-major

    // 0: convert inputs to bf16
    {
        int n = MTOT * DIM;
        cvt_f32_bf16<<<(n / 4 + 255) / 256, 256, 0, stream>>>(x, xb, n);
        n = QKVN * DIM;
        cvt_f32_bf16<<<(n / 4 + 255) / 256, 256, 0, stream>>>(Wqkv, wqb, n);
        n = DIM * DIM;
        cvt_f32_bf16<<<(n / 4 + 255) / 256, 256, 0, stream>>>(Wproj, wpb, n);
    }
    // 1: QKV projection -> q (scaled), k, vT
    qkv_gemm<<<dim3(MTOT / 128, QKVN / 128), 256, 0, stream>>>(xb, wqb, bqkv, q, k, vt);
    // 2: flash attention -> ob
    flash_attn<<<dim3(SEQ / 128, BATCH * NH), 256, 0, stream>>>(q, k, vt, ob);
    // 3: output projection -> out (fp32)
    proj_gemm<<<dim3(MTOT / 128, DIM / 128), 256, 0, stream>>>(ob, wpb, bproj, out);
}
